// Decoder_block_68624987455763
// MI455X (gfx1250) — matrix operand order repaired and checked by reading, not compiled, not run on hardware
//
#include <hip/hip_runtime.h>
#include <hip/hip_bf16.h>

typedef __bf16 bf16_t;
typedef __attribute__((ext_vector_type(16))) __bf16 v16bf;
typedef __attribute__((ext_vector_type(8)))  __bf16 v8bf;
typedef __attribute__((ext_vector_type(8)))  float  v8f;

#define DMODEL 256
#define NHEAD  4
#define DHEAD  64
#define DFF_   1024

// ---------------------------------------------------------------------------
// A-fragment loader for V_WMMA_*_16x16x32 (16-bit A matrix, 16x32 MxK):
//   lane L (row M = L%16, half = L/16):
//     halves 0..7  <- K = k0 + half*16 + h
//     halves 8..15 <- K = k0 + half*16 + h
// Two contiguous 16-byte loads per lane.
// ---------------------------------------------------------------------------
__device__ __forceinline__ v16bf load_a_frag(const bf16_t* rowp, int lh) {
  v8bf lo = *(const v8bf*)(rowp + lh * 16);
  v8bf hi = *(const v8bf*)(rowp + lh * 16 + 8);
  v16bf a;
#pragma unroll
  for (int i = 0; i < 8; ++i) { a[i] = lo[i]; a[8 + i] = hi[i]; }
  return a;
}

// ---------------------------------------------------------------------------
// Elementwise fp32 -> bf16 conversion (and transposing variant for Wo2).
// ---------------------------------------------------------------------------
__global__ void cvt_f32_bf16(const float* __restrict__ s, bf16_t* __restrict__ d, int n) {
  int i = blockIdx.x * blockDim.x + threadIdx.x;
  if (i < n) d[i] = (bf16_t)s[i];
}

__global__ void cvt_f32_bf16_T(const float* __restrict__ s, bf16_t* __restrict__ d,
                               int rows, int cols) {
  int i = blockIdx.x * blockDim.x + threadIdx.x;
  if (i < rows * cols) {
    int r = i / cols, c = i - r * cols;
    d[c * rows + r] = (bf16_t)s[i];   // d[N,K] = s[K,N]^T
  }
}

// ---------------------------------------------------------------------------
// GEMM: C[M,N] = act( A[M,K] @ B[N,K]^T ), bf16 operands, fp32 accumulate.
// Block = 128 threads (4 waves); wave computes a 16x64 C tile; the 64x32 B
// tile is shared by all waves and staged into LDS with the CDNA5 async
// global->LDS path (GLOBAL_LOAD_ASYNC_TO_LDS_B128, ASYNCcnt-tracked),
// double-buffered so the next tile's DMA overlaps this tile's WMMAs.
// B-fragment (32x16 KxN, 16-bit): lane L holds column N = L%16,
//   halves 0..15 <- K = k0 + (L/16)*16 + h  (one contiguous 32B LDS read).
// ---------------------------------------------------------------------------
template <bool RELU, typename OUT_T>
__global__ __launch_bounds__(128) void gemm_wmma(const bf16_t* __restrict__ A,
                                                 const bf16_t* __restrict__ Bm,
                                                 OUT_T* __restrict__ C,
                                                 int M, int N, int K) {
  __shared__ alignas(32) bf16_t Bt[2][64 * 32];   // 2 x 4KB B tiles

  const int wave = threadIdx.x >> 5;
  const int lane = threadIdx.x & 31;
  const int lm = lane & 15, lh = lane >> 4;
  const int row0 = blockIdx.y * 64 + wave * 16;
  const int col0 = blockIdx.x * 64;
  const int t = threadIdx.x;

  v8f acc[4] = {};
  const bf16_t* arow = A + (size_t)(row0 + lm) * K;

  // Stage one 64x32 bf16 B tile into LDS: 256 x 16B segments, 2 per thread,
  // issued as async global->LDS DMA (one instruction per wave per segment).
  auto stage = [&](int buf, int k0) {
#pragma unroll
    for (int s = 0; s < 2; ++s) {
      const int seg = t + s * 128;          // 0..255
      const int row = seg >> 2;             // 0..63
      const int o8  = (seg & 3) * 8;        // bf16 element offset in row
      const bf16_t* gp = Bm + (size_t)(col0 + row) * K + k0 + o8;
      unsigned lds = (unsigned)(size_t)&Bt[buf][row * 32 + o8];
      unsigned long long ga = (unsigned long long)(size_t)gp;
      asm volatile("global_load_async_to_lds_b128 %0, %1, off"
                   :: "v"(lds), "v"(ga)
                   : "memory");
    }
  };

  stage(0, 0);
  for (int k0 = 0; k0 < K; k0 += 32) {
    const int cur = (k0 >> 5) & 1;
    if (k0 + 32 < K) {
      stage(cur ^ 1, k0 + 32);
      // Async loads retire in order: <=2 outstanding means buffer `cur` landed.
      asm volatile("s_wait_asynccnt 0x2" ::: "memory");
    } else {
      asm volatile("s_wait_asynccnt 0x0" ::: "memory");
    }
    __syncthreads();

    v16bf a = load_a_frag(arow + k0, lh);
    if (k0 + 64 < K) __builtin_prefetch(arow + k0 + 64, 0, 0);  // global_prefetch_b8

#pragma unroll
    for (int nt = 0; nt < 4; ++nt) {
      v16bf b = *(const v16bf*)&Bt[cur][(nt * 16 + lm) * 32 + lh * 16];
      acc[nt] = __builtin_amdgcn_wmma_f32_16x16x32_bf16(
          false, a, false, b, (short)0, acc[nt], false, false);
    }
    __syncthreads();   // all reads of Bt[cur] done before it is re-staged
  }

  // C layout: vgpr r, lane L -> row = r + 8*(L/16), col = L%16 (per 16-wide tile)
#pragma unroll
  for (int nt = 0; nt < 4; ++nt)
#pragma unroll
    for (int r = 0; r < 8; ++r) {
      float v = acc[nt][r];
      if (RELU) v = v > 0.f ? v : 0.f;
      const int row = row0 + r + 8 * lh;
      const int col = col0 + nt * 16 + lm;
      C[(size_t)row * N + col] = (OUT_T)v;
    }
}

// ---------------------------------------------------------------------------
// Flash-style attention. Q,K,V are [B*L, D] bf16 with head h at cols h*64..h*64+63.
// Block = 128 threads (4 waves); wave w handles q rows [qblk + 16w, +16).
// Streams 32-key chunks: score WMMAs -> online softmax (shuffle butterflies
// within the C-layout half-waves) -> P via LDS relayout -> P@V WMMAs.
// ---------------------------------------------------------------------------
__global__ __launch_bounds__(128) void attn_wmma(const bf16_t* __restrict__ Q,
                                                 const bf16_t* __restrict__ Kb,
                                                 const bf16_t* __restrict__ Vb,
                                                 const float* __restrict__ mask,
                                                 bf16_t* __restrict__ O,
                                                 int Tq, int Skv, int causal) {
  __shared__ alignas(32) bf16_t Vt[64 * 32];        // V chunk transposed: Vt[n][k]
  __shared__ alignas(32) bf16_t Pbuf[4][16 * 32];   // per-wave P tile (row-major 16x32)

  const int b = blockIdx.z, h = blockIdx.y;
  const int wave = threadIdx.x >> 5;
  const int lane = threadIdx.x & 31;
  const int lm = lane & 15, lh = lane >> 4;
  const int qblk = blockIdx.x * 64;
  const int q0 = qblk + wave * 16;

  // Q fragments for the full 64-wide head dim (two K=32 steps).
  const bf16_t* qrow = Q + (size_t)(b * Tq + q0 + lm) * DMODEL + h * DHEAD;
  v16bf a0 = load_a_frag(qrow, lh);
  v16bf a1 = load_a_frag(qrow + 32, lh);

  float mrow[8], lrow[8];
  v8f oacc[4] = {};
#pragma unroll
  for (int r = 0; r < 8; ++r) { mrow[r] = -1e30f; lrow[r] = 0.f; }

  const int kmax = causal ? (qblk + 64) : Skv;

  for (int kb = 0; kb < kmax; kb += 32) {
    // Cooperative stage of the 32x64 V chunk, transposed into LDS.
    {
      const int tt = threadIdx.x;
      const int k = tt >> 2;             // 0..31
      const int nb = (tt & 3) * 16;      // 0,16,32,48
      const bf16_t* vp = Vb + (size_t)(b * Skv + kb + k) * DMODEL + h * DHEAD + nb;
      v16bf vv = *(const v16bf*)vp;
      if (kb + 32 < kmax) __builtin_prefetch(vp + 32 * DMODEL, 0, 0);
#pragma unroll
      for (int i = 0; i < 16; ++i) Vt[(nb + i) * 32 + k] = vv[i];
    }
    __syncthreads();

    // Scores: two 16x16 output tiles, each accumulated over K=DHEAD=64.
    v8f s0 = {}, s1 = {};
    const int key0 = kb + lm, key1 = kb + 16 + lm;
    {
      const bf16_t* kp0 = Kb + (size_t)(b * Skv + key0) * DMODEL + h * DHEAD + lh * 16;
      const bf16_t* kp1 = Kb + (size_t)(b * Skv + key1) * DMODEL + h * DHEAD + lh * 16;
      v16bf b00 = *(const v16bf*)(kp0);
      v16bf b01 = *(const v16bf*)(kp0 + 32);
      v16bf b10 = *(const v16bf*)(kp1);
      v16bf b11 = *(const v16bf*)(kp1 + 32);
      s0 = __builtin_amdgcn_wmma_f32_16x16x32_bf16(false, a0, false, b00, (short)0, s0, false, false);
      s0 = __builtin_amdgcn_wmma_f32_16x16x32_bf16(false, a1, false, b01, (short)0, s0, false, false);
      s1 = __builtin_amdgcn_wmma_f32_16x16x32_bf16(false, a0, false, b10, (short)0, s1, false, false);
      s1 = __builtin_amdgcn_wmma_f32_16x16x32_bf16(false, a1, false, b11, (short)0, s1, false, false);
    }

    const float mk0 = mask[(size_t)b * Skv + key0];
    const float mk1 = mask[(size_t)b * Skv + key1];

    float p0[8], p1[8];
#pragma unroll
    for (int r = 0; r < 8; ++r) {
      const int qt = q0 + r + 8 * lh;     // this lane's query position for vgpr r
      float f0 = s0[r] * 0.125f;          // 1/sqrt(64)
      float f1 = s1[r] * 0.125f;
      if (mk0 == 0.f || (causal && key0 > qt)) f0 = -1e30f;
      if (mk1 == 0.f || (causal && key1 > qt)) f1 = -1e30f;
      // Row max across the 16 lanes of this half-wave (xor masks stay < 16).
      float cm = fmaxf(f0, f1);
      cm = fmaxf(cm, __shfl_xor(cm, 1));
      cm = fmaxf(cm, __shfl_xor(cm, 2));
      cm = fmaxf(cm, __shfl_xor(cm, 4));
      cm = fmaxf(cm, __shfl_xor(cm, 8));
      const float mnew = fmaxf(mrow[r], cm);
      const float sc = __expf(mrow[r] - mnew);
      mrow[r] = mnew;
      f0 = __expf(f0 - mnew);
      f1 = __expf(f1 - mnew);
      float rs = f0 + f1;
      rs += __shfl_xor(rs, 1);
      rs += __shfl_xor(rs, 2);
      rs += __shfl_xor(rs, 4);
      rs += __shfl_xor(rs, 8);
      lrow[r] = lrow[r] * sc + rs;
#pragma unroll
      for (int nt = 0; nt < 4; ++nt) oacc[nt][r] *= sc;
      p0[r] = f0; p1[r] = f1;
    }

    // C-layout P -> row-major LDS -> A-fragment layout (wave-local).
    bf16_t* pw = &Pbuf[wave][0];
#pragma unroll
    for (int r = 0; r < 8; ++r) {
      const int pr = r + 8 * lh;
      pw[pr * 32 + lm]      = (bf16_t)p0[r];
      pw[pr * 32 + 16 + lm] = (bf16_t)p1[r];
    }
    asm volatile("s_wait_dscnt 0" ::: "memory");
    v16bf pa = load_a_frag(&Pbuf[wave][lm * 32], lh);

    // O += P(16x32) @ Vchunk(32x64): B-frags from transposed LDS (contiguous).
#pragma unroll
    for (int nt = 0; nt < 4; ++nt) {
      v16bf vv = *(const v16bf*)&Vt[(nt * 16 + lm) * 32 + lh * 16];
      oacc[nt] = __builtin_amdgcn_wmma_f32_16x16x32_bf16(
          false, pa, false, vv, (short)0, oacc[nt], false, false);
    }
    __syncthreads();
  }

#pragma unroll
  for (int nt = 0; nt < 4; ++nt)
#pragma unroll
    for (int r = 0; r < 8; ++r) {
      const int qt = q0 + r + 8 * lh;
      const float inv = 1.f / fmaxf(lrow[r], 1e-20f);
      O[(size_t)(b * Tq + qt) * DMODEL + h * DHEAD + nt * 16 + lm] =
          (bf16_t)(oacc[nt][r] * inv);
    }
}

// ---------------------------------------------------------------------------
// Fused residual + LayerNorm over D=256: one wave per row, 8 rows per block.
// outf = resid + LN(x)*g + b ; optional bf16 mirror for the next GEMM.
// ---------------------------------------------------------------------------
__global__ __launch_bounds__(256) void ln_residual(const float* __restrict__ x,
                                                   const float* __restrict__ resid,
                                                   const float* __restrict__ g,
                                                   const float* __restrict__ bt,
                                                   float* __restrict__ outf,
                                                   bf16_t* __restrict__ outb) {
  const int lane = threadIdx.x & 31;
  const int row = blockIdx.x * 8 + (threadIdx.x >> 5);
  const float* xr = x + (size_t)row * DMODEL;

  float v[8]; float s = 0.f;
#pragma unroll
  for (int i = 0; i < 8; ++i) { v[i] = xr[i * 32 + lane]; s += v[i]; }
  s += __shfl_xor(s, 1); s += __shfl_xor(s, 2); s += __shfl_xor(s, 4);
  s += __shfl_xor(s, 8); s += __shfl_xor(s, 16);
  const float mu = s * (1.f / DMODEL);

  float vs = 0.f;
#pragma unroll
  for (int i = 0; i < 8; ++i) { float dd = v[i] - mu; vs += dd * dd; }
  vs += __shfl_xor(vs, 1); vs += __shfl_xor(vs, 2); vs += __shfl_xor(vs, 4);
  vs += __shfl_xor(vs, 8); vs += __shfl_xor(vs, 16);
  const float rstd = rsqrtf(vs * (1.f / DMODEL) + 1e-5f);

#pragma unroll
  for (int i = 0; i < 8; ++i) {
    const int col = i * 32 + lane;
    const float y = (v[i] - mu) * rstd * g[col] + bt[col];
    const float o = resid[(size_t)row * DMODEL + col] + y;
    outf[(size_t)row * DMODEL + col] = o;
    if (outb) outb[(size_t)row * DMODEL + col] = (bf16_t)o;
  }
}

// ---------------------------------------------------------------------------
// Orchestration.
// ---------------------------------------------------------------------------
extern "C" void kernel_launch(void* const* d_in, const int* in_sizes, int n_in,
                              void* d_out, int out_size, void* d_ws, size_t ws_size,
                              hipStream_t stream) {
  (void)in_sizes; (void)n_in; (void)out_size; (void)ws_size;

  const float* Xt    = (const float*)d_in[0];
  const float* oMask = (const float*)d_in[1];
  const float* Xe    = (const float*)d_in[2];
  const float* iMask = (const float*)d_in[3];
  const float* Wq1   = (const float*)d_in[4];
  const float* Wk1   = (const float*)d_in[5];
  const float* Wv1   = (const float*)d_in[6];
  const float* Wo1   = (const float*)d_in[7];
  const float* Wq2   = (const float*)d_in[8];
  const float* Wk2   = (const float*)d_in[9];
  const float* Wv2   = (const float*)d_in[10];
  const float* Wo2   = (const float*)d_in[11];
  const float* W1    = (const float*)d_in[12];
  const float* W2    = (const float*)d_in[13];
  const float* g1    = (const float*)d_in[14];
  const float* b1    = (const float*)d_in[15];
  const float* g2    = (const float*)d_in[16];
  const float* b2    = (const float*)d_in[17];
  const float* g3    = (const float*)d_in[18];
  const float* b3    = (const float*)d_in[19];

  const int Bb = 16, T = 1024, S = 1024;
  const int NR = Bb * T;  // 16384 rows

  char* ws = (char*)d_ws;
  size_t off = 0;
  auto alloc = [&](size_t bytes) -> void* {
    void* p = ws + off;
    off += (bytes + 255) & ~(size_t)255;
    return p;
  };

  bf16_t* XtB  = (bf16_t*)alloc((size_t)NR * DMODEL * 2);
  bf16_t* XeB  = (bf16_t*)alloc((size_t)NR * DMODEL * 2);
  bf16_t* Qb   = (bf16_t*)alloc((size_t)NR * DMODEL * 2);
  bf16_t* Kbuf = (bf16_t*)alloc((size_t)NR * DMODEL * 2);
  bf16_t* Vbuf = (bf16_t*)alloc((size_t)NR * DMODEL * 2);
  bf16_t* AOb  = (bf16_t*)alloc((size_t)NR * DMODEL * 2);
  bf16_t* Wq1B = (bf16_t*)alloc((size_t)DMODEL * DMODEL * 2);
  bf16_t* Wk1B = (bf16_t*)alloc((size_t)DMODEL * DMODEL * 2);
  bf16_t* Wv1B = (bf16_t*)alloc((size_t)DMODEL * DMODEL * 2);
  bf16_t* Wo1B = (bf16_t*)alloc((size_t)DMODEL * DMODEL * 2);
  bf16_t* Wq2B = (bf16_t*)alloc((size_t)DMODEL * DMODEL * 2);
  bf16_t* Wk2B = (bf16_t*)alloc((size_t)DMODEL * DMODEL * 2);
  bf16_t* Wv2B = (bf16_t*)alloc((size_t)DMODEL * DMODEL * 2);
  bf16_t* Wo2TB= (bf16_t*)alloc((size_t)DMODEL * DMODEL * 2);
  bf16_t* W1B  = (bf16_t*)alloc((size_t)DFF_ * DMODEL * 2);
  bf16_t* W2B  = (bf16_t*)alloc((size_t)DMODEL * DFF_ * 2);
  float*  Of   = (float*)alloc((size_t)NR * DMODEL * 4);
  float*  X1f  = (float*)alloc((size_t)NR * DMODEL * 4);
  bf16_t* X1b  = (bf16_t*)alloc((size_t)NR * DMODEL * 2);
  float*  X2f  = (float*)alloc((size_t)NR * DMODEL * 4);
  bf16_t* X2b  = (bf16_t*)alloc((size_t)NR * DMODEL * 2);
  bf16_t* Hb   = (bf16_t*)alloc((size_t)NR * DFF_ * 2);
  float*  Ff   = (float*)alloc((size_t)NR * DMODEL * 4);

  auto cvt = [&](const float* s, bf16_t* d, int n) {
    cvt_f32_bf16<<<dim3((n + 255) / 256), dim3(256), 0, stream>>>(s, d, n);
  };
  cvt(Xt, XtB, NR * DMODEL);
  cvt(Xe, XeB, NR * DMODEL);
  cvt(Wq1, Wq1B, DMODEL * DMODEL);
  cvt(Wk1, Wk1B, DMODEL * DMODEL);
  cvt(Wv1, Wv1B, DMODEL * DMODEL);
  cvt(Wo1, Wo1B, DMODEL * DMODEL);
  cvt(Wq2, Wq2B, DMODEL * DMODEL);
  cvt(Wk2, Wk2B, DMODEL * DMODEL);
  cvt(Wv2, Wv2B, DMODEL * DMODEL);
  // O @ Wo2 (untransposed in the reference) -> pre-transpose so the GEMM is A@B^T.
  cvt_f32_bf16_T<<<dim3((DMODEL * DMODEL + 255) / 256), dim3(256), 0, stream>>>(
      Wo2, Wo2TB, DMODEL, DMODEL);
  cvt(W1, W1B, DFF_ * DMODEL);
  cvt(W2, W2B, DMODEL * DFF_);

  const dim3 blk(128);
  const dim3 gD(DMODEL / 64, NR / 64);   // N=256 outputs
  const dim3 gF(DFF_ / 64, NR / 64);     // N=1024 outputs
  const dim3 gA(T / 64, NHEAD, Bb);

  // ---- Layer 1: masked self-attention ----
  gemm_wmma<false, bf16_t><<<gD, blk, 0, stream>>>(XtB, Wq1B, Qb,   NR, DMODEL, DMODEL);
  gemm_wmma<false, bf16_t><<<gD, blk, 0, stream>>>(XtB, Wk1B, Kbuf, NR, DMODEL, DMODEL);
  gemm_wmma<false, bf16_t><<<gD, blk, 0, stream>>>(XtB, Wv1B, Vbuf, NR, DMODEL, DMODEL);
  attn_wmma<<<gA, blk, 0, stream>>>(Qb, Kbuf, Vbuf, oMask, AOb, T, T, 1);
  gemm_wmma<false, float><<<gD, blk, 0, stream>>>(AOb, Wo1B, Of, NR, DMODEL, DMODEL);
  ln_residual<<<dim3(NR / 8), dim3(256), 0, stream>>>(Of, Xt, g1, b1, X1f, X1b);

  // ---- Layer 2: cross-attention ----
  gemm_wmma<false, bf16_t><<<gD, blk, 0, stream>>>(X1b, Wq2B, Qb,   NR, DMODEL, DMODEL);
  gemm_wmma<false, bf16_t><<<gD, blk, 0, stream>>>(XeB, Wk2B, Kbuf, NR, DMODEL, DMODEL);
  gemm_wmma<false, bf16_t><<<gD, blk, 0, stream>>>(XeB, Wv2B, Vbuf, NR, DMODEL, DMODEL);
  attn_wmma<<<gA, blk, 0, stream>>>(Qb, Kbuf, Vbuf, iMask, AOb, T, S, 0);
  gemm_wmma<false, float><<<gD, blk, 0, stream>>>(AOb, Wo2TB, Of, NR, DMODEL, DMODEL);
  ln_residual<<<dim3(NR / 8), dim3(256), 0, stream>>>(Of, X1f, g2, b2, X2f, X2b);

  // ---- FFN ----
  gemm_wmma<true, bf16_t><<<gF, blk, 0, stream>>>(X2b, W1B, Hb, NR, DFF_, DMODEL);
  gemm_wmma<true, float><<<gD, blk, 0, stream>>>(Hb, W2B, Ff, NR, DMODEL, DFF_);
  ln_residual<<<dim3(NR / 8), dim3(256), 0, stream>>>(Ff, X2f, g3, b3,
                                                      (float*)d_out, nullptr);
}
